// RNN_Autoencoder_71588514890536
// MI455X (gfx1250) — matrix operand order repaired and checked by reading, not compiled, not run on hardware
//
#include <hip/hip_runtime.h>

// ===========================================================================
// RNN autoencoder (LSTM enc -> LSTM dec -> tanh projection) for gfx1250.
//
// Single PERSISTENT kernel runs all 256 sequential timesteps internally,
// synchronized with a device-scope atomic grid barrier (release/acquire
// fences -> global_wb / global_inv on CDNA5).  This removes ~390 kernel
// launches whose dispatch latency would otherwise dominate (each step's GEMM
// is ~2 GFLOP => sub-microsecond on bf16 WMMA, launches are ~1-2us each).
//
// GEMM core: v_wmma_f32_16x16x32_bf16, fragments loaded straight from
// row-major global memory with two 128-bit loads per lane (K = 16g + e per lane half g,
// same order in A and B); weights stay L2-resident (~7.3 MiB << 192 MB).
//
// __launch_bounds__(256, 2): force the compiler to fit >=2 workgroups per CU
// so the persistent grid (128 blocks) is guaranteed launchable/co-resident
// regardless of the physical VGPR file size (spin-barrier safety).
// Workspace requirement: ~42 MiB.
// ===========================================================================

typedef __bf16 bf16;
typedef __attribute__((ext_vector_type(16))) __bf16 v16bf;
typedef __attribute__((ext_vector_type(8)))  __bf16 v8bf;
typedef __attribute__((ext_vector_type(8)))  float  v8f;

#define B_ 512
#define T_ 128
#define E_ 256
#define H_ 512
#define Z_ 256
#define NB_ 128          // persistent grid: (B/64) x (H/32) = 8 x 16 blocks

// ---------------------------------------------------------------------------
// A fragment: 16x32 (MxK) bf16, row-major source, leading dim lda.
//   lane: m = lane&15, g = lane>>4;  half e holds K = 16g + e:
//   16 contiguous K elements: two 16B loads.
// ---------------------------------------------------------------------------
__device__ __forceinline__ v16bf load_a_frag(const bf16* base, int lda, int lane) {
  const int m = lane & 15;
  const int g = lane >> 4;
  const bf16* p = base + (size_t)m * lda + 16 * g;
  v8bf lo = *reinterpret_cast<const v8bf*>(p);
  v8bf hi = *reinterpret_cast<const v8bf*>(p + 8);
  v16bf r;
#pragma unroll
  for (int i = 0; i < 8; ++i) { r[i] = lo[i]; r[i + 8] = hi[i]; }
  return r;
}

// ---------------------------------------------------------------------------
// B fragment: 32x16 (KxN) where B = W^T, W row-major [N,K] (PyTorch layout).
//   lane: n = lane&15, g = lane>>4; half e holds K = 16g + e: the 16 halves
//   are 16 contiguous K elements of row n of W -> two 16B loads.
// ---------------------------------------------------------------------------
__device__ __forceinline__ v16bf load_b_fragT(const bf16* base, int ldw, int lane) {
  const int n = lane & 15;
  const int g = lane >> 4;
  const bf16* p = base + (size_t)n * ldw + 16 * g;
  v8bf lo = *reinterpret_cast<const v8bf*>(p);
  v8bf hi = *reinterpret_cast<const v8bf*>(p + 8);
  v16bf r;
#pragma unroll
  for (int i = 0; i < 8; ++i) { r[i] = lo[i]; r[i + 8] = hi[i]; }
  return r;
}

__device__ __forceinline__ v8f wmma_bf16(v16bf a, v16bf b, v8f c) {
  return __builtin_amdgcn_wmma_f32_16x16x32_bf16(
      false, a, false, b, (short)0, c, false, false);
}

__device__ __forceinline__ float fast_sigmoid(float x) {
  return 1.0f / (1.0f + __expf(-x));
}

// ---------------------------------------------------------------------------
// Device-wide barrier for the persistent kernel.  Every wave drains/releases
// its own stores (global_wb at DEV scope), block arrives via ACQ_REL RMW,
// last block bumps the generation; waiters spin with s_sleep; ACQUIRE fence
// on exit invalidates WGP$ so next step sees the new h written by other WGPs.
// ---------------------------------------------------------------------------
__device__ __forceinline__ void grid_barrier(unsigned* bar, unsigned* gen) {
  __builtin_amdgcn_fence(__ATOMIC_RELEASE, "agent");
  __syncthreads();
  if (threadIdx.x == 0) {
    unsigned g = __hip_atomic_load(gen, __ATOMIC_RELAXED, __HIP_MEMORY_SCOPE_AGENT);
    unsigned arrived =
        __hip_atomic_fetch_add(bar, 1u, __ATOMIC_ACQ_REL, __HIP_MEMORY_SCOPE_AGENT);
    if (arrived == (unsigned)(NB_ - 1)) {
      __hip_atomic_store(bar, 0u, __ATOMIC_RELAXED, __HIP_MEMORY_SCOPE_AGENT);
      __hip_atomic_store(gen, g + 1u, __ATOMIC_RELEASE, __HIP_MEMORY_SCOPE_AGENT);
    } else {
      while (__hip_atomic_load(gen, __ATOMIC_ACQUIRE, __HIP_MEMORY_SCOPE_AGENT) == g)
        __builtin_amdgcn_s_sleep(1);
    }
  }
  __syncthreads();
  __builtin_amdgcn_fence(__ATOMIC_ACQUIRE, "agent");
}

// ---------------------------------------------------------------------------
// One fused LSTM step for this block's tile (64 batch rows x 32 hidden cols,
// all 4 gates).  8 waves: wg = w>>1 (gate i,f,g,o), wm = w&1 (row half);
// each wave 32x32 -> 2x2 f32 accumulators.  Gates exchanged through LDS,
// then c' = sig(f)*c + sig(i)*tanh(g), h' = sig(o)*tanh(c').
// ---------------------------------------------------------------------------
__device__ __forceinline__ void lstm_tile(
    const bf16* __restrict__ A1, int lda1, int K1, const bf16* __restrict__ W1,
    const bf16* __restrict__ A2, int K2, const bf16* __restrict__ W2,  // lda2 = H_
    const float* __restrict__ bias,
    float* __restrict__ c, bf16* __restrict__ h_out,
    int m0, int h0, float (*gsm)[64][33])
{
  const int tid  = threadIdx.x;
  const int lane = tid & 31;
  const int w    = tid >> 5;
  const int wg   = w >> 1;
  const int wm   = w & 1;
  const int mw   = m0 + wm * 32;
  const int ncol = wg * H_ + h0;

  v8f acc[2][2];
#pragma unroll
  for (int in = 0; in < 2; ++in) {
    const float bv = bias[ncol + in * 16 + (lane & 15)];
#pragma unroll
    for (int im = 0; im < 2; ++im)
#pragma unroll
      for (int r = 0; r < 8; ++r) acc[im][in][r] = bv;
  }

#pragma unroll
  for (int phase = 0; phase < 2; ++phase) {
    const bf16* A   = phase ? A2 : A1;
    const bf16* Wt  = phase ? W2 : W1;
    const int   lda = phase ? H_ : lda1;
    const int   K   = phase ? K2 : K1;
    for (int k = 0; k < K; k += 32) {
      const bf16* b0p = Wt + (size_t)ncol * K + k;
      if (k + 64 < K) __builtin_prefetch(b0p + 64, 0, 1);  // global_prefetch_b8
      v16bf a0 = load_a_frag(A + (size_t)mw * lda + k, lda, lane);
      v16bf a1 = load_a_frag(A + (size_t)(mw + 16) * lda + k, lda, lane);
      v16bf b0 = load_b_fragT(b0p, K, lane);
      v16bf b1 = load_b_fragT(Wt + (size_t)(ncol + 16) * K + k, K, lane);
      acc[0][0] = wmma_bf16(a0, b0, acc[0][0]);
      acc[1][0] = wmma_bf16(a1, b0, acc[1][0]);
      acc[0][1] = wmma_bf16(a0, b1, acc[0][1]);
      acc[1][1] = wmma_bf16(a1, b1, acc[1][1]);
    }
  }

  // Gate nonlinearity + LDS exchange.  C-frag: VGPR r -> M = 8*(lane>>4)+r,
  // N = lane&15.
#pragma unroll
  for (int im = 0; im < 2; ++im)
#pragma unroll
    for (int in = 0; in < 2; ++in)
#pragma unroll
      for (int r = 0; r < 8; ++r) {
        const int ml = wm * 32 + im * 16 + 8 * (lane >> 4) + r;
        const int nl = in * 16 + (lane & 15);
        float v = acc[im][in][r];
        v = (wg == 2) ? tanhf(v) : fast_sigmoid(v);
        gsm[wg][ml][nl] = v;
      }
  __syncthreads();

  const int ml = tid >> 2;
  const int nb = (tid & 3) * 8;
#pragma unroll
  for (int j = 0; j < 8; ++j) {
    const int nl = nb + j;
    const float iv = gsm[0][ml][nl];
    const float fv = gsm[1][ml][nl];
    const float gv = gsm[2][ml][nl];
    const float ov = gsm[3][ml][nl];
    const size_t gi = (size_t)(m0 + ml) * H_ + (h0 + nl);
    const float cn = fv * c[gi] + iv * gv;
    c[gi] = cn;
    h_out[gi] = (bf16)(ov * tanhf(cn));
  }
}

// ---------------------------------------------------------------------------
// z tile: out[:, t, :] slice = tanh(h @ zW^T + zb).  Executed by blocks 0..31
// (tile 64x64 of the 512x256 output; 8 waves, each 32x16 -> 2 accumulators).
// ---------------------------------------------------------------------------
__device__ __forceinline__ void z_tile(
    const bf16* __restrict__ h, const bf16* __restrict__ zW,
    const float* __restrict__ zb, float* __restrict__ out, int t)
{
  const int tid  = threadIdx.x;
  const int lane = tid & 31;
  const int w    = tid >> 5;
  const int wm   = w & 1;
  const int wn   = w >> 1;                       // 0..3
  const int mw   = (blockIdx.x & 7) * 64 + wm * 32;
  const int nc   = (blockIdx.x >> 3) * 64 + wn * 16;   // blockIdx < 32 -> 0..3

  v8f acc[2];
  const float bv = zb[nc + (lane & 15)];
#pragma unroll
  for (int im = 0; im < 2; ++im)
#pragma unroll
    for (int r = 0; r < 8; ++r) acc[im][r] = bv;

  for (int k = 0; k < H_; k += 32) {
    v16bf a0 = load_a_frag(h + (size_t)mw * H_ + k, H_, lane);
    v16bf a1 = load_a_frag(h + (size_t)(mw + 16) * H_ + k, H_, lane);
    v16bf b  = load_b_fragT(zW + (size_t)nc * H_ + k, H_, lane);
    acc[0] = wmma_bf16(a0, b, acc[0]);
    acc[1] = wmma_bf16(a1, b, acc[1]);
  }

#pragma unroll
  for (int im = 0; im < 2; ++im)
#pragma unroll
    for (int r = 0; r < 8; ++r) {
      const int row = mw + im * 16 + 8 * (lane >> 4) + r;
      const int col = nc + (lane & 15);
      out[(size_t)row * (T_ * Z_) + (size_t)t * Z_ + col] = tanhf(acc[im][r]);
    }
}

// ---------------------------------------------------------------------------
// Persistent kernel: all 128 encoder + 128 decoder steps, grid-barriered.
// Block -> fixed (m0, h0) gate tile.  h ping-pongs between hb0/hb1; c is
// updated in place (each element owned by a fixed block).  Decoder iteration
// t also computes z_{t-1} on blocks 0..31 (reads hcur, which is only
// overwritten in iteration t+1, i.e. after the barrier).
// ---------------------------------------------------------------------------
__global__ __launch_bounds__(256, 2) void rnn_persistent_kernel(
    const bf16* __restrict__ x_bf,
    const bf16* __restrict__ ewih, const bf16* __restrict__ ewhh,
    const float* __restrict__ ebias,
    const bf16* __restrict__ dwih, const bf16* __restrict__ dwhh,
    const float* __restrict__ dbias,
    const bf16* __restrict__ zw, const float* __restrict__ zb,
    bf16* __restrict__ hb0, bf16* __restrict__ hb1,
    const bf16* __restrict__ cin, float* __restrict__ cst,
    float* __restrict__ out,
    unsigned* bar, unsigned* gen)
{
  __shared__ float gsm[4][64][33];     // 33-padded gate exchange buffer
  const int m0 = (blockIdx.x & 7) * 64;
  const int h0 = (blockIdx.x >> 3) * 32;
  bf16* hb[2] = { hb0, hb1 };

  // ---- encoder ----
  for (int t = 0; t < T_; ++t) {
    lstm_tile(x_bf + (size_t)t * E_, T_ * E_, E_, ewih,
              hb[t & 1], H_, ewhh, ebias, cst, hb[(t + 1) & 1],
              m0, h0, gsm);
    grid_barrier(bar, gen);
  }
  // final encoder h is in hb[T_ & 1] == hb[0]; ping-pong parity continues.

  // ---- decoder ----
  for (int t = 0; t < T_; ++t) {
    const bf16* hcur = hb[t & 1];
    if (t > 0 && blockIdx.x < 32)
      z_tile(hcur, zw, zb, out, t - 1);          // z_{t-1} = tanh(h_{t-1} @ zW^T + zb)
    const bf16* a1 = (t == 0) ? cin : hcur;      // step-0 input is const 0.1
    lstm_tile(a1, H_, H_, dwih,
              hcur, H_, dwhh, dbias, cst, hb[(t + 1) & 1],
              m0, h0, gsm);
    grid_barrier(bar, gen);
  }
  if (blockIdx.x < 32)
    z_tile(hb[T_ & 1], zw, zb, out, T_ - 1);
}

// ------------------------- prep / utility kernels --------------------------
__global__ void f2bf_kernel(const float* __restrict__ s, bf16* __restrict__ d, int n) {
  for (int i = blockIdx.x * blockDim.x + threadIdx.x; i < n; i += gridDim.x * blockDim.x)
    d[i] = (bf16)s[i];
}
__global__ void fill_bf_kernel(bf16* __restrict__ d, float v, int n) {
  const bf16 bv = (bf16)v;
  for (int i = blockIdx.x * blockDim.x + threadIdx.x; i < n; i += gridDim.x * blockDim.x)
    d[i] = bv;
}
__global__ void fill_f_kernel(float* __restrict__ d, float v, int n) {
  for (int i = blockIdx.x * blockDim.x + threadIdx.x; i < n; i += gridDim.x * blockDim.x)
    d[i] = v;
}
__global__ void add_bias_kernel(const float* __restrict__ a, const float* __restrict__ b,
                                float* __restrict__ o, int n) {
  for (int i = blockIdx.x * blockDim.x + threadIdx.x; i < n; i += gridDim.x * blockDim.x)
    o[i] = a[i] + b[i];
}
__global__ void zero_sync_kernel(unsigned* bar, unsigned* gen) {
  if (threadIdx.x == 0) { *bar = 0u; *gen = 0u; }
}

// ===========================================================================
extern "C" void kernel_launch(void* const* d_in, const int* in_sizes, int n_in,
                              void* d_out, int out_size, void* d_ws, size_t ws_size,
                              hipStream_t stream) {
  (void)in_sizes; (void)n_in; (void)out_size; (void)ws_size;
  const float* x    = (const float*)d_in[0];   // [B,T,E]
  const float* eWih = (const float*)d_in[1];   // [4H,E]
  const float* eWhh = (const float*)d_in[2];   // [4H,H]
  const float* ebih = (const float*)d_in[3];
  const float* ebhh = (const float*)d_in[4];
  const float* dWih = (const float*)d_in[5];   // [4H,H]
  const float* dWhh = (const float*)d_in[6];   // [4H,H]
  const float* dbih = (const float*)d_in[7];
  const float* dbhh = (const float*)d_in[8];
  const float* zW   = (const float*)d_in[9];   // [Z,H]
  const float* zb   = (const float*)d_in[10];  // [Z]
  float* out = (float*)d_out;                  // [B,T,Z]

  // ---- workspace carve-up (~42 MiB) ----
  char* p = (char*)d_ws;
  auto carve = [&p](size_t bytes) {
    char* r = p;
    p += (bytes + 255) & ~(size_t)255;
    return r;
  };
  bf16*  x_bf   = (bf16*)carve((size_t)B_ * T_ * E_ * 2);
  bf16*  ewih_b = (bf16*)carve((size_t)4 * H_ * E_ * 2);
  bf16*  ewhh_b = (bf16*)carve((size_t)4 * H_ * H_ * 2);
  bf16*  dwih_b = (bf16*)carve((size_t)4 * H_ * H_ * 2);
  bf16*  dwhh_b = (bf16*)carve((size_t)4 * H_ * H_ * 2);
  bf16*  zw_b   = (bf16*)carve((size_t)Z_ * H_ * 2);
  float* ebias  = (float*)carve((size_t)4 * H_ * 4);
  float* dbias  = (float*)carve((size_t)4 * H_ * 4);
  bf16*  hbuf0  = (bf16*)carve((size_t)B_ * H_ * 2);
  bf16*  hbuf1  = (bf16*)carve((size_t)B_ * H_ * 2);
  bf16*  cin    = (bf16*)carve((size_t)B_ * H_ * 2);
  float* cst    = (float*)carve((size_t)B_ * H_ * 4);
  unsigned* syncmem = (unsigned*)carve(256);   // [0]=bar, [1]=gen

  // ---- prep (deterministic per call; resets all persistent-kernel state) ----
  f2bf_kernel<<<2048, 256, 0, stream>>>(x, x_bf, B_ * T_ * E_);
  f2bf_kernel<<<512, 256, 0, stream>>>(eWih, ewih_b, 4 * H_ * E_);
  f2bf_kernel<<<512, 256, 0, stream>>>(eWhh, ewhh_b, 4 * H_ * H_);
  f2bf_kernel<<<512, 256, 0, stream>>>(dWih, dwih_b, 4 * H_ * H_);
  f2bf_kernel<<<512, 256, 0, stream>>>(dWhh, dwhh_b, 4 * H_ * H_);
  f2bf_kernel<<<256, 256, 0, stream>>>(zW, zw_b, Z_ * H_);
  add_bias_kernel<<<8, 256, 0, stream>>>(ebih, ebhh, ebias, 4 * H_);
  add_bias_kernel<<<8, 256, 0, stream>>>(dbih, dbhh, dbias, 4 * H_);
  fill_bf_kernel<<<256, 256, 0, stream>>>(hbuf0, 0.1f, B_ * H_);   // h0 = 0.1
  fill_bf_kernel<<<256, 256, 0, stream>>>(cin, 0.1f, B_ * H_);     // decoder const input
  fill_f_kernel<<<256, 256, 0, stream>>>(cst, 0.1f, B_ * H_);      // c0 = 0.1
  zero_sync_kernel<<<1, 32, 0, stream>>>(syncmem, syncmem + 1);

  // ---- the whole network: one persistent launch ----
  rnn_persistent_kernel<<<NB_, 256, 0, stream>>>(
      x_bf, ewih_b, ewhh_b, ebias, dwih_b, dwhh_b, dbias, zw_b, zb,
      hbuf0, hbuf1, cin, cst, out, syncmem, syncmem + 1);
}
